// StandardAttention_7756710936884
// MI455X (gfx1250) — matrix operand order repaired and checked by reading, not compiled, not run on hardware
//
#include <hip/hip_runtime.h>

// ---------------- problem constants (reference setup_inputs) ----------------
#define B_   4
#define S_   2048
#define D_   2048
#define H_   16
#define DK_  128            // D_/H_
#define M_   8192           // B_*S_
#define SCALE_ 0.08838834764831845f  // 1/sqrt(128)
#define NEG_INF_ (-1e30f)

typedef __attribute__((ext_vector_type(8)))  float    v8f;
typedef __attribute__((ext_vector_type(16))) __bf16   v16bf;
typedef __attribute__((ext_vector_type(8)))  __bf16   v8bf;
typedef __attribute__((ext_vector_type(4)))  unsigned u32x4;
typedef __attribute__((ext_vector_type(4)))  int      i32x4;
typedef __attribute__((ext_vector_type(8)))  int      i32x8;

#define WMMA_BF16(a, b, c) \
  __builtin_amdgcn_wmma_f32_16x16x32_bf16(false, (a), false, (b), (short)0, (c), false, false)

// A-fragment (16x32 bf16): lane<16 holds row=lane, K {0..7}U{16..23};
// lane>=16 holds row=lane-16, K {8..15}U{24..31}. Two b128 loads.
__device__ __forceinline__ v16bf load_a_frag(const __bf16* base, int stride, int lane) {
  int row = lane & 15;
  int kb  = (lane >> 4) << 3;
  const __bf16* p = base + (size_t)row * stride + kb;
  v8bf lo = *(const v8bf*)(p);
  v8bf hi = *(const v8bf*)(p + 16);
  return __builtin_shufflevector(lo, hi, 0,1,2,3,4,5,6,7,8,9,10,11,12,13,14,15);
}

// B-fragment (32x16 bf16) from B^T stored [N][K] row-major:
// lane<16: col=lane, K {0..7}U{16..23}; lane>=16: col=lane-16, K {8..15}U{24..31}. Two b128 loads.
__device__ __forceinline__ v16bf load_b_frag(const __bf16* base, int stride, int lane) {
  int row = lane & 15;
  int kb  = (lane >> 4) << 3;
  const __bf16* p = base + (size_t)row * stride + kb;
  v8bf lo = *(const v8bf*)(p);
  v8bf hi = *(const v8bf*)(p + 16);
  return __builtin_shufflevector(lo, hi, 0,1,2,3,4,5,6,7,8,9,10,11,12,13,14,15);
}

// ---- Tensor Data Mover: 2D tile (32 elems x 128 rows, bf16) global -> LDS ----
// D# built per CDNA5 ISA ch.8: group0 = {count/type/lds_addr/global_addr},
// group1 = {data_size, tensor dims, tile dims, stride}. Groups 2/3 unused (2D).
__device__ __forceinline__ void tdm_load_tile_128x32(unsigned lds_addr,
                                                     const __bf16* gptr,
                                                     unsigned stride_elems) {
  unsigned long long ga = (unsigned long long)(uintptr_t)gptr;  // byte address
  u32x4 g0;
  g0[0] = 1u;                                   // count=1 (valid descriptor)
  g0[1] = lds_addr;                             // LDS byte address
  g0[2] = (unsigned)(ga & 0xffffffffu);         // global_addr[31:0]
  g0[3] = (unsigned)((ga >> 32) & 0x01ffffffu)  // global_addr[56:32]
          | 0x80000000u;                        // type=2 ("image") in bits[127:126]
  const unsigned TD0 = stride_elems;            // tensor_dim0 (elements, in-bounds)
  const unsigned TD1 = 0x000fffffu;             // tensor_dim1 (large; tiles in-bounds)
  i32x8 g1;
  g1[0] = (int)(1u << 16);                      // data_size=1 -> 2 bytes/elem
  g1[1] = (int)((TD0 & 0xffffu) << 16);         // tensor_dim0[15:0] @ bits63:48
  g1[2] = (int)(((TD0 >> 16) & 0xffffu) | ((TD1 & 0xffffu) << 16));
  g1[3] = (int)(((TD1 >> 16) & 0xffffu) | (32u << 16));   // tile_dim0 = 32
  g1[4] = (int)(128u);                          // tile_dim1 = 128, tile_dim2 = 0
  g1[5] = (int)stride_elems;                    // tensor_dim0_stride[31:0]
  g1[6] = 0;                                    // stride hi + dim1_stride lo
  g1[7] = 0;
  i32x4 z4 = {0, 0, 0, 0};
#if defined(__clang_major__) && (__clang_major__ >= 23)
  i32x8 z8 = {0, 0, 0, 0, 0, 0, 0, 0};
  __builtin_amdgcn_tensor_load_to_lds(g0, g1, z4, z4, z8, 0);
#else
  __builtin_amdgcn_tensor_load_to_lds(g0, g1, z4, z4, 0);
#endif
}

// ---------------- conversion kernels ----------------
__global__ void cvt_bf16(const float* __restrict__ x, __bf16* __restrict__ y, int n) {
  int i = blockIdx.x * blockDim.x + threadIdx.x;
  if (i < n) y[i] = (__bf16)x[i];
}

// W [K][N] fp32 -> Wt [N][K] bf16 (coalesced writes)
__global__ void cvt_transpose(const float* __restrict__ W, __bf16* __restrict__ Wt, int K, int N) {
  size_t idx = (size_t)blockIdx.x * blockDim.x + threadIdx.x;
  if (idx >= (size_t)K * N) return;
  int n = (int)(idx / K);
  int k = (int)(idx % K);
  Wt[idx] = (__bf16)W[(size_t)k * N + n];
}

// ------------- GEMM: C[M][N] = A[M][K] @ Bt[N][K]^T, TDM-staged LDS -------------
// Workgroup (8 waves) computes a 128x128 tile. Per 32-deep k-step, wave 0 DMAs
// A(128x32) and Bt(128x32) into double-buffered LDS via the Tensor Data Mover;
// each wave computes a 32x64 sub-tile (2 A-frags x 4 B-frags = 8 WMMA).
template <bool OUT_BF16>
__global__ __launch_bounds__(256) void gemm_bf16(const __bf16* __restrict__ A,
                                                 const __bf16* __restrict__ Bt,
                                                 void* __restrict__ C,
                                                 int M, int N, int K) {
  __shared__ __bf16 sA[2][128 * 32];
  __shared__ __bf16 sB[2][128 * 32];

  const int lane = threadIdx.x & 31;
  const int wid  = threadIdx.x >> 5;
  const int ntb  = N >> 7;                 // 128-wide column tiles
  const int mt   = blockIdx.x / ntb;
  const int nt   = blockIdx.x % ntb;
  const int mw   = (wid & 3) * 32;         // wave row offset within 128
  const int nw   = (wid >> 2) * 64;        // wave col offset within 128

  const __bf16* Ab = A  + (size_t)mt * 128 * K;
  const __bf16* Bb = Bt + (size_t)nt * 128 * K;
  const unsigned ldsA = (unsigned)(uintptr_t)&sA[0][0];
  const unsigned ldsB = (unsigned)(uintptr_t)&sB[0][0];
  const unsigned bufBytes = 128 * 32 * 2;

  const int niter = K >> 5;

  // prologue: stage k-step 0 into buffer 0
  if (wid == 0) {
    tdm_load_tile_128x32(ldsA, Ab, (unsigned)K);
    tdm_load_tile_128x32(ldsB, Bb, (unsigned)K);
  }

  v8f acc[2][4] = {};
  int cur = 0;
  for (int it = 0; it < niter; ++it) {
    if (wid == 0) __builtin_amdgcn_s_wait_tensorcnt(0);
    __syncthreads();                              // buf[cur] ready for all waves
    if (wid == 0 && (it + 1) < niter) {           // prefetch next k-step via TDM
      int k0 = (it + 1) << 5;
      tdm_load_tile_128x32(ldsA + (unsigned)(cur ^ 1) * bufBytes, Ab + k0, (unsigned)K);
      tdm_load_tile_128x32(ldsB + (unsigned)(cur ^ 1) * bufBytes, Bb + k0, (unsigned)K);
    }
    const __bf16* a_tile = &sA[cur][0];
    const __bf16* b_tile = &sB[cur][0];
    v16bf a0 = load_a_frag(a_tile + (size_t)(mw + 0)  * 32, 32, lane);
    v16bf a1 = load_a_frag(a_tile + (size_t)(mw + 16) * 32, 32, lane);
#pragma unroll
    for (int j = 0; j < 4; ++j) {
      v16bf b = load_b_frag(b_tile + (size_t)(nw + j * 16) * 32, 32, lane);
      acc[0][j] = WMMA_BF16(a0, b, acc[0][j]);
      acc[1][j] = WMMA_BF16(a1, b, acc[1][j]);
    }
    cur ^= 1;
  }

  const int l16 = lane >> 4, colk = lane & 15;
#pragma unroll
  for (int f = 0; f < 2; ++f)
#pragma unroll
    for (int j = 0; j < 4; ++j)
#pragma unroll
      for (int i = 0; i < 8; ++i) {
        size_t r = (size_t)mt * 128 + mw + f * 16 + i + 8 * l16;
        size_t c = (size_t)nt * 128 + nw + j * 16 + colk;
        if (OUT_BF16) ((__bf16*)C)[r * N + c] = (__bf16)acc[f][j][i];
        else          ((float*)C)[r * N + c]  = acc[f][j][i];
      }
}

// ---------------- rotary + head split (tables indexed by HEAD per reference) ----
// qkv [B,S,3,H,dk] bf16 -> Qh,Kh [B,H,S,dk] (rotated), Vt [B,H,dk,S]
__global__ void rotary_split(const __bf16* __restrict__ qkv,
                             __bf16* __restrict__ Qh, __bf16* __restrict__ Kh,
                             __bf16* __restrict__ Vt) {
  int idx = blockIdx.x * blockDim.x + threadIdx.x;   // B*H*S*64 threads
  int d = idx & 63;
  int s = (idx >> 6) & (S_ - 1);
  int h = (idx >> 17) & (H_ - 1);
  int b = idx >> 21;

  float c, sn;
  if (d < 32) {
    float f  = __powf(10000.0f, -(float)d * (1.0f / 31.0f));
    float th = (float)h * f;
    c = __cosf(th); sn = __sinf(th);
  } else { c = 1.0f; sn = 0.0f; }

  size_t rowoff = ((size_t)(b * S_ + s)) * (3 * D_);
  size_t qkoff  = ((size_t)(b * H_ + h) * S_ + s) * DK_ + d;

  {
    float x1 = (float)qkv[rowoff + 0 * D_ + h * DK_ + d];
    float x2 = (float)qkv[rowoff + 0 * D_ + h * DK_ + d + 64];
    Qh[qkoff]      = (__bf16)(x1 * c + x2 * sn);
    Qh[qkoff + 64] = (__bf16)(-x1 * sn + x2 * c);
  }
  {
    float x1 = (float)qkv[rowoff + 1 * D_ + h * DK_ + d];
    float x2 = (float)qkv[rowoff + 1 * D_ + h * DK_ + d + 64];
    Kh[qkoff]      = (__bf16)(x1 * c + x2 * sn);
    Kh[qkoff + 64] = (__bf16)(-x1 * sn + x2 * c);
  }
  {
    size_t voff = ((size_t)(b * H_ + h) * DK_ + d) * S_ + s;
    Vt[voff]                   = (__bf16)(float)qkv[rowoff + 2 * D_ + h * DK_ + d];
    Vt[voff + (size_t)64 * S_] = (__bf16)(float)qkv[rowoff + 2 * D_ + h * DK_ + d + 64];
  }
}

// ---------------- flash attention: one wave per 16-row query tile ----------------
__global__ __launch_bounds__(256) void flash_attn(const __bf16* __restrict__ Qh,
                                                  const __bf16* __restrict__ Kh,
                                                  const __bf16* __restrict__ Vt,
                                                  __bf16* __restrict__ Oh) {
  __shared__ __bf16 smem[8][16 * 32];

  const int lane = threadIdx.x & 31;
  const int wid  = threadIdx.x >> 5;
  const int qt   = blockIdx.x * 8 + wid;
  const int bh   = blockIdx.y;           // b*H + h
  const int q0   = qt * 16;
  const int l16  = lane >> 4;
  const int colk = lane & 15;

  const __bf16* Qb = Qh + ((size_t)bh * S_ + q0) * DK_;
  const __bf16* Kb = Kh + (size_t)bh * S_ * DK_;
  const __bf16* Vb = Vt + (size_t)bh * DK_ * S_;

  v16bf q[4];
#pragma unroll
  for (int t = 0; t < 4; ++t) q[t] = load_a_frag(Qb + 32 * t, DK_, lane);

  v8f acc[8] = {};
  float m[8], l[8];
#pragma unroll
  for (int i = 0; i < 8; ++i) { m[i] = NEG_INF_; l[i] = 0.0f; }

  const int kv_end = (((q0 + 47) >> 5) << 5);   // ceil((q0+16)/32)*32
  for (int kv = 0; kv < kv_end; kv += 32) {
    // ---- S = Q @ K^T for 32 keys (two 16-col fragments) ----
    v8f s0 = {}, s1 = {};
    const __bf16* K0 = Kb + (size_t)kv * DK_;
    const __bf16* K1 = Kb + (size_t)(kv + 16) * DK_;
    __builtin_prefetch(Kb + (size_t)(kv + 32) * DK_ + lane * 16, 0, 0);  // next chunk
#pragma unroll
    for (int t = 0; t < 4; ++t) {
      v16bf b0 = load_b_frag(K0 + 32 * t, DK_, lane);
      s0 = WMMA_BF16(q[t], b0, s0);
      v16bf b1 = load_b_frag(K1 + 32 * t, DK_, lane);
      s1 = WMMA_BF16(q[t], b1, s1);
    }

    // ---- causal mask + online softmax (row reductions across 16 lanes) ----
#pragma unroll
    for (int i = 0; i < 8; ++i) {
      int qr = q0 + i + 8 * l16;
      float v0 = (kv + colk      <= qr) ? s0[i] * SCALE_ : NEG_INF_;
      float v1 = (kv + 16 + colk <= qr) ? s1[i] * SCALE_ : NEG_INF_;
      float rm = fmaxf(v0, v1);
#pragma unroll
      for (int off = 1; off < 16; off <<= 1) rm = fmaxf(rm, __shfl_xor(rm, off, 16));
      float mn = fmaxf(m[i], rm);
      float al = __expf(m[i] - mn);
      float p0 = __expf(v0 - mn);
      float p1 = __expf(v1 - mn);
      float rs = p0 + p1;
#pragma unroll
      for (int off = 1; off < 16; off <<= 1) rs += __shfl_xor(rs, off, 16);
      l[i] = l[i] * al + rs;
      m[i] = mn;
#pragma unroll
      for (int j = 0; j < 8; ++j) acc[j][i] = acc[j][i] * al;
      smem[wid][(i + 8 * l16) * 32 + colk]      = (__bf16)p0;
      smem[wid][(i + 8 * l16) * 32 + colk + 16] = (__bf16)p1;
    }

    // ---- C-layout -> A-layout via LDS (in-wave DS ops are ordered) ----
    __builtin_amdgcn_wave_barrier();
    v16bf pa;
    {
      int row = lane & 15;
      int kb  = l16 * 8;
      const __bf16* sp = &smem[wid][row * 32 + kb];
#pragma unroll
      for (int jj = 0; jj < 8; ++jj) { pa[jj] = sp[jj]; pa[8 + jj] = sp[16 + jj]; }
    }
    __builtin_amdgcn_wave_barrier();

    // ---- O += P @ V (V already transposed: rows are head dims) ----
#pragma unroll
    for (int j = 0; j < 8; ++j) {
      v16bf vb = load_b_frag(Vb + (size_t)(16 * j) * S_ + kv, S_, lane);
      acc[j] = WMMA_BF16(pa, vb, acc[j]);
    }
  }

  // ---- normalize and store to Oh [B,S,D] bf16 ----
  const int b = bh >> 4, h = bh & (H_ - 1);
#pragma unroll
  for (int i = 0; i < 8; ++i) {
    float inv = 1.0f / l[i];
    int s = q0 + i + 8 * l16;
    size_t base = ((size_t)(b * S_ + s)) * D_ + h * DK_;
#pragma unroll
    for (int j = 0; j < 8; ++j)
      Oh[base + 16 * j + colk] = (__bf16)(acc[j][i] * inv);
  }
}

// ---------------- host launch ----------------
extern "C" void kernel_launch(void* const* d_in, const int* in_sizes, int n_in,
                              void* d_out, int out_size, void* d_ws, size_t ws_size,
                              hipStream_t stream) {
  (void)in_sizes; (void)n_in; (void)out_size; (void)ws_size;
  const float* x    = (const float*)d_in[0];
  const float* Wqkv = (const float*)d_in[1];
  const float* Wo   = (const float*)d_in[2];
  float* out = (float*)d_out;

  char*  ws  = (char*)d_ws;
  size_t off = 0;
  auto alloc = [&](size_t bytes) -> void* {
    void* p = ws + off;
    off += (bytes + 255) & ~(size_t)255;
    return p;
  };

  __bf16* x_h   = (__bf16*)alloc((size_t)M_ * D_ * 2);        //  33.5 MB
  __bf16* WqT   = (__bf16*)alloc((size_t)3 * D_ * D_ * 2);    //  25.2 MB
  __bf16* WoT   = (__bf16*)alloc((size_t)D_ * D_ * 2);        //   8.4 MB
  __bf16* qkv_h = (__bf16*)alloc((size_t)M_ * 3 * D_ * 2);    // 100.7 MB
  __bf16* Qh    = (__bf16*)alloc((size_t)M_ * D_ * 2);        //  33.5 MB
  __bf16* Kh    = (__bf16*)alloc((size_t)M_ * D_ * 2);        //  33.5 MB
  __bf16* Vt    = (__bf16*)alloc((size_t)M_ * D_ * 2);        //  33.5 MB
  __bf16* Oh    = (__bf16*)alloc((size_t)M_ * D_ * 2);        //  33.5 MB

  // 1) convert inputs to bf16 (weights transposed for B-fragment loads)
  cvt_bf16<<<(M_ * D_) / 256, 256, 0, stream>>>(x, x_h, M_ * D_);
  cvt_transpose<<<(3 * D_ * D_) / 256, 256, 0, stream>>>(Wqkv, WqT, D_, 3 * D_);
  cvt_transpose<<<(D_ * D_) / 256, 256, 0, stream>>>(Wo, WoT, D_, D_);

  // 2) QKV projection: [8192 x 6144] = [8192 x 2048] @ [2048 x 6144]
  gemm_bf16<true><<<(M_ / 128) * (3 * D_ / 128), 256, 0, stream>>>(
      x_h, WqT, qkv_h, M_, 3 * D_, D_);

  // 3) rotary (head-indexed tables per reference) + head split + V transpose
  rotary_split<<<(B_ * H_ * S_ * 64) / 256, 256, 0, stream>>>(qkv_h, Qh, Kh, Vt);

  // 4) causal flash attention
  flash_attn<<<dim3(S_ / 16 / 8, B_ * H_), 256, 0, stream>>>(Qh, Kh, Vt, Oh);

  // 5) output projection: out = Oh @ Wo (fp32 result)
  gemm_bf16<false><<<(M_ / 128) * (D_ / 128), 256, 0, stream>>>(
      Oh, WoT, out, M_, D_, D_);
}
